// DTNHybridFFN_23295902613827
// MI455X (gfx1250) — matrix operand order repaired and checked by reading, not compiled, not run on hardware
//
#include <hip/hip_runtime.h>
#include <hip/hip_fp16.h>

typedef __attribute__((ext_vector_type(16))) _Float16 v16h;
typedef __attribute__((ext_vector_type(8)))  _Float16 v8h;
typedef __attribute__((ext_vector_type(8)))  float    v8f;
typedef __attribute__((ext_vector_type(4)))  float    v4f;
typedef __attribute__((ext_vector_type(4)))  unsigned int v4u;
typedef __attribute__((ext_vector_type(8)))  int      v8i;
typedef __attribute__((ext_vector_type(4)))  int      v4i;

#define T_TOK 32768   // B*S
#define DM    256     // D_MODEL
#define FF    1024    // FFN

// -----------------------------------------------------------------------------
// Pack W [K][N] f32 row-major into WMMA B-fragment order, f16.
// dst index = (((nt*(K/32) + kt)*32 + lane)*16 + j)
// -----------------------------------------------------------------------------
__global__ void pack_b_f16(const float* __restrict__ W, _Float16* __restrict__ out,
                           int K, int N) {
  int tid = blockIdx.x * blockDim.x + threadIdx.x;
  int total = K * N;
  if (tid >= total) return;
  int j  = tid & 15;
  int l  = (tid >> 4) & 31;
  int kt = (tid >> 9) % (K >> 5);
  int nt = tid / ((K >> 5) << 9);
  int n  = nt * 16 + (l & 15);
  int k  = kt * 32 + ((j >> 3) << 4) + ((l >> 4) << 3) + (j & 7);
  out[tid] = (_Float16)W[k * N + n];
}

// -----------------------------------------------------------------------------
// TDM descriptor: DMA a 64x64 f32 tile (row stride 256 elems) from global
// into LDS, padded +1 DWORD every 64 DWORDs -> 65-float LDS row stride
// (bank-conflict-free column reads).  ISA cdna5 §8 D# layout, 2D tensor.
// -----------------------------------------------------------------------------
__device__ __forceinline__ void tdm_load_wt_tile(const float* gsrc, unsigned int lds_off) {
  unsigned long long ga = (unsigned long long)(uintptr_t)gsrc;
  v4u g0;
  g0[0] = 1u;                                              // count=1, user mode
  g0[1] = lds_off;                                         // lds_addr
  g0[2] = (unsigned int)ga;                                // global_addr[31:0]
  g0[3] = (unsigned int)((ga >> 32) & 0x01FFFFFFu)         // global_addr[56:32]
        | (2u << 30);                                      // type=2 (image)
  v8i g1;
  g1[0] = (2 << 16)      // data_size = 4 bytes
        | (1 << 20)      // pad_enable
        | (5 << 22);     // pad_interval: 64 DWORDs
                         // pad_amount=0 -> 1 DWORD;  workgroup_mask=0
  g1[1] = (int)(256u << 16);    // tensor_dim0 = 256  (bits 79:48)
  g1[2] = (int)(1024u << 16);   // tensor_dim1 = 1024 (bits 111:80)
  g1[3] = (int)(64u << 16);     // tile_dim0 = 64     (bits 127:112)
  g1[4] = 64;                   // tile_dim1 = 64, tile_dim2 = 0
  g1[5] = 256;                  // tensor_dim0_stride = 256 (low 32)
  g1[6] = 0;                    // stride hi / dim1_stride lo
  g1[7] = 0;
  v4i z4 = {0, 0, 0, 0};
  v8i z8 = {0, 0, 0, 0, 0, 0, 0, 0};
  __builtin_amdgcn_tensor_load_to_lds(g0, g1, z4, z4, z8, 0);
}

// -----------------------------------------------------------------------------
// Pass 1: per 16-token x 64-FFN tile (4 waves, one 16x16 tile each):
//   tropical max-plus core (VALU) with TDM double-buffered Wt staging,
//   x@Wc and x@Wg via WMMA f16, fused epilogue -> f16.
// -----------------------------------------------------------------------------
__global__ __launch_bounds__(128)
void ffn_pass1(const float* __restrict__ x,   const float* __restrict__ Wt,
               const float* __restrict__ bt,
               const float* __restrict__ sl_cvx, const float* __restrict__ of_cvx,
               const float* __restrict__ sl_ccv, const float* __restrict__ of_ccv,
               const float* __restrict__ alpha,
               const _Float16* __restrict__ pWc, const float* __restrict__ bc,
               const _Float16* __restrict__ pWg, const float* __restrict__ bg,
               _Float16* __restrict__ fused)
{
  __shared__ __align__(16) float sX[DM][20];          // x transposed [d][m], padded
  __shared__ __align__(32) _Float16 sA[8][32][16];    // WMMA A fragments (f16)
  __shared__ __align__(16) float sWd[2][64 * 65];     // TDM-padded Wt chunks [n][65]

  const int tid  = threadIdx.x;
  const int lane = tid & 31;
  const int w    = tid >> 5;
  const int t0   = blockIdx.x * 16;
  const int f0   = blockIdx.y * 64;

  // Kick off TDM load of Wt chunk 0 (overlaps with x staging below).
  if (w == 0) {
    tdm_load_wt_tile(Wt + (size_t)f0 * DM,
                     (unsigned int)(uintptr_t)&sWd[0][0]);
  }

  // Stage x tile: sX[d][m] (f32 for tropical) + sA (f16 WMMA A layout).
  for (int idx = tid; idx < 16 * DM; idx += 128) {
    int d = idx & (DM - 1);
    int m = idx >> 8;
    float xv = x[(size_t)(t0 + m) * DM + d];
    sX[d][m] = xv;
    int kt = d >> 5, ko = d & 31;
    int hi = (ko >> 3) & 1;                  // A interleave per ISA 7.12.2
    int j  = (ko < 8) ? ko : (ko < 16 ? ko - 8 : (ko < 24 ? ko - 8 : ko - 16));
    sA[kt][m + (hi ? 16 : 0)][j] = (_Float16)xv;
  }

  const int nloc = (w << 4) + (lane & 15);
  const int m0   = (lane < 16) ? 0 : 8;      // matches WMMA C fragment rows

  float acc[8];
  #pragma unroll
  for (int j = 0; j < 8; ++j) acc[j] = -3.402823466e38f;

  // Tropical core: d in 4 chunks of 64, TDM double-buffered Wt staging.
  for (int dc = 0; dc < 4; ++dc) {
    if (w == 0) __builtin_amdgcn_s_wait_tensorcnt(0);   // chunk dc landed in LDS
    __syncthreads();                                    // publish to all waves
    if (w == 0 && dc < 3) {                             // prefetch chunk dc+1
      tdm_load_wt_tile(Wt + (size_t)f0 * DM + (size_t)(dc + 1) * 64,
                       (unsigned int)(uintptr_t)&sWd[(dc + 1) & 1][0]);
    }
    const float* wrow = &sWd[dc & 1][0] + nloc * 65;
    #pragma unroll 4
    for (int dl = 0; dl < 64; ++dl) {
      int d = dc * 64 + dl;
      float wt = wrow[dl];
      v4f xa = *(const v4f*)&sX[d][m0];
      v4f xb = *(const v4f*)&sX[d][m0 + 4];
      #pragma unroll
      for (int j = 0; j < 4; ++j) {
        acc[j]     = fmaxf(acc[j],     xa[j] + wt);
        acc[j + 4] = fmaxf(acc[j + 4], xb[j] + wt);
      }
    }
  }

  // WMMA: classical and gate GEMM fragments (K = 256 -> 8 steps each).
  v8f accC = {};
  v8f accG = {};
  const int ft = blockIdx.y * 4 + w;
  const v16h* bC = (const v16h*)pWc + (size_t)(ft * 8) * 32 + lane;
  const v16h* bG = (const v16h*)pWg + (size_t)(ft * 8) * 32 + lane;
  #pragma unroll
  for (int kt = 0; kt < 8; ++kt) {
    v16h a  = *(const v16h*)&sA[kt][lane][0];
    v16h b0 = bC[kt * 32];
    v16h b1 = bG[kt * 32];
    accC = __builtin_amdgcn_wmma_f32_16x16x32_f16(false, a, false, b0, (short)0, accC, false, false);
    accG = __builtin_amdgcn_wmma_f32_16x16x32_f16(false, a, false, b1, (short)0, accG, false, false);
  }

  // Epilogue (all per-lane: column f fixed, 8 token rows in fragment).
  const int f = f0 + nloc;
  const float btv = bt[f], bcv = bc[f], bgv = bg[f];
  const float s = 1.0f / (1.0f + __expf(-alpha[f]));
  float slx[8], ofx[8], slv[8], ofv[8];
  #pragma unroll
  for (int p = 0; p < 8; ++p) {
    slx[p] = sl_cvx[f * 8 + p]; ofx[p] = of_cvx[f * 8 + p];
    slv[p] = sl_ccv[f * 8 + p]; ofv[p] = of_ccv[f * 8 + p];
  }
  #pragma unroll
  for (int j = 0; j < 8; ++j) {
    float tt  = acc[j] + btv;
    float cvx = -3.402823466e38f, ccv = 3.402823466e38f;
    #pragma unroll
    for (int p = 0; p < 8; ++p) {
      cvx = fmaxf(cvx, tt * slx[p] + ofx[p]);
      ccv = fminf(ccv, tt * slv[p] + ofv[p]);
    }
    float trop = s * cvx + (1.0f - s) * ccv;
    float pre  = accC[j] + bcv;
    float gel  = 0.5f * pre * (1.0f + erff(pre * 0.70710678118654752f));
    float gpre = accG[j] + bgv;
    float g    = 1.0f / (1.0f + __expf(-gpre));
    float fu   = g * trop + (1.0f - g) * gel;
    fused[(size_t)(t0 + m0 + j) * FF + f] = (_Float16)fu;
  }
}

// -----------------------------------------------------------------------------
// Pass 2: out = fused(f16) @ Wd + bd, pure WMMA (K = 1024 -> 32 steps).
// -----------------------------------------------------------------------------
__global__ __launch_bounds__(128)
void ffn_pass2(const _Float16* __restrict__ fused, const _Float16* __restrict__ pWd,
               const float* __restrict__ bd, float* __restrict__ out)
{
  const int tid  = threadIdx.x;
  const int lane = tid & 31;
  const int w    = tid >> 5;
  const int t0   = blockIdx.x * 16;
  const int nt   = blockIdx.y * 4 + w;
  const int row  = t0 + (lane & 15);        // A-fragment: lane holds row l%16
  const int ho   = (lane >= 16) ? 8 : 0;    // K-interleave offset for this half
  const _Float16* arow = fused + (size_t)row * FF;
  const v16h* bW = (const v16h*)pWd + (size_t)nt * 32 * 32 + lane;

  v8f acc = {};
  #pragma unroll 4
  for (int kt = 0; kt < 32; ++kt) {
    union { v16h v; struct { v8h l8; v8h h8; } s; } a;
    a.s.l8 = *(const v8h*)&arow[kt * 32 + ho];
    a.s.h8 = *(const v8h*)&arow[kt * 32 + 16 + ho];
    v16h b = bW[kt * 32];
    acc = __builtin_amdgcn_wmma_f32_16x16x32_f16(false, a.v, false, b, (short)0, acc, false, false);
  }

  const int n = nt * 16 + (lane & 15);
  const float bdv = bd[n];
  #pragma unroll
  for (int j = 0; j < 8; ++j)
    out[(size_t)(t0 + ho + j) * DM + n] = acc[j] + bdv;
}

// -----------------------------------------------------------------------------
extern "C" void kernel_launch(void* const* d_in, const int* in_sizes, int n_in,
                              void* d_out, int out_size, void* d_ws, size_t ws_size,
                              hipStream_t stream) {
  (void)in_sizes; (void)n_in; (void)out_size; (void)ws_size;
  const float* x      = (const float*)d_in[0];
  const float* Wt     = (const float*)d_in[1];
  const float* bt     = (const float*)d_in[2];
  const float* sl_cvx = (const float*)d_in[3];
  const float* of_cvx = (const float*)d_in[4];
  const float* sl_ccv = (const float*)d_in[5];
  const float* of_ccv = (const float*)d_in[6];
  const float* alpha  = (const float*)d_in[7];
  const float* Wc     = (const float*)d_in[8];
  const float* bc     = (const float*)d_in[9];
  const float* Wg     = (const float*)d_in[10];
  const float* bg     = (const float*)d_in[11];
  const float* Wd     = (const float*)d_in[12];
  const float* bd     = (const float*)d_in[13];
  float* out = (float*)d_out;

  char* ws = (char*)d_ws;
  _Float16* fused = (_Float16*)ws;                                  // 64 MB
  _Float16* pWc   = (_Float16*)(ws + (size_t)T_TOK * FF * 2);       // 512 KB
  _Float16* pWg   = pWc + (size_t)DM * FF;
  _Float16* pWd   = pWg + (size_t)DM * FF;

  const int nPack = DM * FF;                 // 262144 elements each
  pack_b_f16<<<nPack / 256, 256, 0, stream>>>(Wc, pWc, DM, FF);
  pack_b_f16<<<nPack / 256, 256, 0, stream>>>(Wg, pWg, DM, FF);
  pack_b_f16<<<nPack / 256, 256, 0, stream>>>(Wd, pWd, FF, DM);

  dim3 g1(T_TOK / 16, FF / 64);
  ffn_pass1<<<g1, 128, 0, stream>>>(x, Wt, bt, sl_cvx, of_cvx, sl_ccv, of_ccv,
                                    alpha, pWc, bc, pWg, bg, fused);
  dim3 g2(T_TOK / 16, DM / 64);
  ffn_pass2<<<g2, 128, 0, stream>>>(fused, pWd, bd, out);
}
